// StableDiagonalSSM_21715354648867
// MI455X (gfx1250) — matrix operand order repaired and checked by reading, not compiled, not run on hardware
//
#include <hip/hip_runtime.h>
#include <hip/hip_bf16.h>
#include <stdint.h>

typedef __attribute__((ext_vector_type(16))) __bf16 v16bf;
typedef __attribute__((ext_vector_type(8)))  float  v8f;

union FragB16 { uint4 q[2]; v16bf v; };

__device__ __forceinline__ unsigned short f2bf(float f) {
    unsigned int u = __float_as_uint(f);
    unsigned int r = u + 0x7FFFu + ((u >> 16) & 1u);   // round-to-nearest-even
    return (unsigned short)(r >> 16);
}

__device__ __forceinline__ float softplus_f(float x) {
    return (x > 20.0f) ? x : __logf(1.0f + __expf(x));
}

// gfx1250 async global->LDS copy (ASYNCcnt-tracked), 16 bytes per lane.
__device__ __forceinline__ void async_load_b128(unsigned lds_addr, const void* gaddr) {
    asm volatile("global_load_async_to_lds_b128 %0, %1, off"
                 :: "v"(lds_addr), "v"(gaddr) : "memory");
}
__device__ __forceinline__ void wait_async0() {
    asm volatile("s_wait_asynccnt 0x0" ::: "memory");
}

// ---------------------------------------------------------------------------
// f32 -> bf16 conversion (grid-stride)
// ---------------------------------------------------------------------------
__global__ void cvt_bf16_kernel(const float* __restrict__ src,
                                unsigned short* __restrict__ dst, int n) {
    for (int i = blockIdx.x * blockDim.x + threadIdx.x; i < n;
         i += gridDim.x * blockDim.x)
        dst[i] = f2bf(src[i]);
}

// ---------------------------------------------------------------------------
// WMMA bf16 GEMM:  C[M,N] = A[M,K] * W[N,K]^T   with fused mode epilogues
// block = 256 threads (8 waves). Block tile: 256(M) x 64(N).
// Each wave: 32(M) x 64(N) -> 8 accumulators; each B fragment feeds 2 WMMAs.
// All 4 B fragments are loaded into distinct registers before the WMMA burst
// so ds_load latency is hidden behind earlier matrix ops (staggered dscnt).
// B tile (64 x 32 bf16 = 4KB) staged in LDS via async copy, double buffered.
// A fragments double-buffered in registers.
// ---------------------------------------------------------------------------
struct GemmArgs {
    const unsigned short* A;   // bf16 [M x K]
    const unsigned short* W;   // bf16 [N x K]
    int K;
    int mode;
    // mode 0: in-proj -> u (bf16) + silu(gate) (f32)
    unsigned short* out_u;
    float*          out_sg;
    // mode 1: state/dt proj -> bT, decay  transposed [B*128, 2048]
    float*          out_bT;
    float*          out_dec;
    const float*    b_dt;
    const float*    A_log;
    // mode 2: C-proj -> z = y*sg + x*D  (bf16)
    unsigned short* out_z;
    const float*    sg;
    const float*    x;
    const float*    Dv;
    // mode 3: out-proj -> f32 output
    float*          out_f32;
};

__global__ __launch_bounds__(256) void gemm_bf16_kernel(GemmArgs g) {
    // Two 4KB buffers: 64 rows (n) x 64 bytes (32 bf16 of K)
    __shared__ __align__(16) unsigned char Bsm[2][64 * 64];

    const int tid  = threadIdx.x;
    const int lane = tid & 31;
    const int wave = tid >> 5;
    const int hi   = lane >> 4;       // half-wave selector
    const int lo   = lane & 15;
    const int rb   = blockIdx.y * 256 + wave * 32;   // wave M base (32 rows)
    const int cb   = blockIdx.x * 64;                // block N base
    const int K    = g.K;

    v8f acc[2][4] = {};

    const unsigned short* Arow0 = g.A + (size_t)(rb + lo) * K;
    const unsigned short* Arow1 = g.A + (size_t)(rb + 16 + lo) * K;

    // async stage of B tile for a given kb into buffer `buf`
    const int srow = tid >> 2;        // 0..63  (n within tile)
    const int sch  = tid & 3;         // 16B chunk within 64B row
    auto stageB = [&](int kb, int buf) {
        const unsigned short* gp = g.W + (size_t)(cb + srow) * K + kb + sch * 8;
        unsigned lds = (unsigned)(uintptr_t)&Bsm[buf][srow * 64 + sch * 16];
        async_load_b128(lds, gp);
    };

    // ---- prologue: stage first B tile, load first A fragments ----
    stageB(0, 0);
    FragB16 aCur0, aCur1;
    aCur0.q[0] = *(const uint4*)(Arow0 + 8 * hi);
    aCur0.q[1] = *(const uint4*)(Arow0 + 16 + 8 * hi);
    aCur1.q[0] = *(const uint4*)(Arow1 + 8 * hi);
    aCur1.q[1] = *(const uint4*)(Arow1 + 16 + 8 * hi);
    wait_async0();
    __syncthreads();

    for (int kb = 0; kb < K; kb += 32) {
        const int  cur      = (kb >> 5) & 1;
        const bool has_next = (kb + 32) < K;

        // kick off next stage (B -> LDS async, A -> regs)
        if (has_next) stageB(kb + 32, cur ^ 1);
        FragB16 aNext0, aNext1;
        if (has_next) {
            aNext0.q[0] = *(const uint4*)(Arow0 + kb + 32 + 8 * hi);
            aNext0.q[1] = *(const uint4*)(Arow0 + kb + 48 + 8 * hi);
            aNext1.q[0] = *(const uint4*)(Arow1 + kb + 32 + 8 * hi);
            aNext1.q[1] = *(const uint4*)(Arow1 + kb + 48 + 8 * hi);
        }

        // read ALL B fragments first (distinct registers -> staggered waits)
        const unsigned char* bs = Bsm[cur];
        FragB16 bf[4];
#pragma unroll
        for (int t = 0; t < 4; ++t) {
            const unsigned char* bp = bs + (t * 16 + lo) * 64 + 16 * hi;
            bf[t].q[0] = *(const uint4*)(bp);
            bf[t].q[1] = *(const uint4*)(bp + 32);
        }
        // WMMA burst: each B fragment feeds both M sub-tiles
#pragma unroll
        for (int t = 0; t < 4; ++t) {
            acc[0][t] = __builtin_amdgcn_wmma_f32_16x16x32_bf16(
                false, aCur0.v, false, bf[t].v, (short)0, acc[0][t], false, false);
            acc[1][t] = __builtin_amdgcn_wmma_f32_16x16x32_bf16(
                false, aCur1.v, false, bf[t].v, (short)0, acc[1][t], false, false);
        }

        wait_async0();          // next B tile landed (issued at top)
        __syncthreads();        // all waves done reading `cur`, next buf ready
        if (has_next) { aCur0 = aNext0; aCur1 = aNext1; }
    }

    // Epilogue. C/D layout: VGPR v -> row v + 8*hi, column = lo.
#pragma unroll
    for (int s = 0; s < 2; ++s) {
#pragma unroll
        for (int t = 0; t < 4; ++t) {
            const int c = cb + t * 16 + lo;
#pragma unroll
            for (int v = 0; v < 8; ++v) {
                const int r   = rb + s * 16 + v + 8 * hi;
                const float val = acc[s][t][v];
                if (g.mode == 0) {
                    if (c < 1024) {
                        g.out_u[(size_t)r * 1024 + c] = f2bf(val);
                    } else {
                        float sv = val / (1.0f + __expf(-val));   // silu(gate)
                        g.out_sg[(size_t)r * 1024 + (c - 1024)] = sv;
                    }
                } else if (g.mode == 1) {
                    const int b = r >> 11, l = r & 2047;
                    if (c < 128) {
                        g.out_bT[((size_t)(b * 128 + c)) * 2048 + l] = val;
                    } else {
                        const int n = c - 128;
                        float dt   = fminf(softplus_f(val + g.b_dt[n]), 10.0f);
                        float rate = softplus_f(g.A_log[n]) + 1e-4f;
                        float dec  = __expf(-dt * rate);
                        dec = fminf(fmaxf(dec, 0.0f), 1.0f);
                        g.out_dec[((size_t)(b * 128 + n)) * 2048 + l] = dec;
                    }
                } else if (g.mode == 2) {
                    const size_t idx = (size_t)r * 1024 + c;
                    float z = val * g.sg[idx] + g.x[idx] * g.Dv[c];
                    g.out_z[idx] = f2bf(z);
                } else {
                    g.out_f32[(size_t)r * 1024 + c] = val;
                }
            }
        }
    }
}

// ---------------------------------------------------------------------------
// Affine scan:  h_t = d_t * h_{t-1} + (1-d_t) * b_t over L=2048.
// One block per (b,n) pair (512 blocks). 256 threads x 8 steps each.
// Segment carry (P,S): h_end = P*h_start + S ; Hillis-Steele scan in LDS.
// Inputs are transposed [B*128, 2048]; output hs is [B*2048, 128] bf16.
// ---------------------------------------------------------------------------
__global__ __launch_bounds__(256) void scan_kernel(
        const float* __restrict__ dec, const float* __restrict__ bT,
        unsigned short* __restrict__ hs) {
    __shared__ float sP[256];
    __shared__ float sS[256];
    const int bn  = blockIdx.x;          // 0..511  -> (b = bn>>7, n = bn&127)
    const int tid = threadIdx.x;
    const size_t base = (size_t)bn * 2048 + tid * 8;

    float dv[8], cv[8];
    float P = 1.0f, S = 0.0f;
#pragma unroll
    for (int i = 0; i < 8; ++i) {
        dv[i] = dec[base + i];
        cv[i] = (1.0f - dv[i]) * bT[base + i];
        P = dv[i] * P;
        S = dv[i] * S + cv[i];
    }
    sP[tid] = P; sS[tid] = S;
    __syncthreads();

    for (int off = 1; off < 256; off <<= 1) {
        float myP = sP[tid], myS = sS[tid];
        float pP = 1.0f, pS = 0.0f;
        if (tid >= off) { pP = sP[tid - off]; pS = sS[tid - off]; }
        __syncthreads();
        sP[tid] = myP * pP;            // earlier-then-me composition
        sS[tid] = myP * pS + myS;
        __syncthreads();
    }

    float h = (tid == 0) ? 0.0f : sS[tid - 1];   // h0 = 0 -> carry = S only
    const int b = bn >> 7, n = bn & 127;
    size_t obase = ((size_t)b * 2048 + tid * 8) * 128 + n;
#pragma unroll
    for (int i = 0; i < 8; ++i) {
        h = dv[i] * h + cv[i];
        hs[obase + (size_t)i * 128] = f2bf(h);
    }
}

// ---------------------------------------------------------------------------
extern "C" void kernel_launch(void* const* d_in, const int* in_sizes, int n_in,
                              void* d_out, int out_size, void* d_ws, size_t ws_size,
                              hipStream_t stream) {
    const float* x       = (const float*)d_in[0];  // [4,2048,1024]
    const float* W_in    = (const float*)d_in[1];  // [2048,1024]
    const float* W_state = (const float*)d_in[2];  // [128,1024]
    const float* W_dt    = (const float*)d_in[3];  // [128,1024]
    const float* b_dt    = (const float*)d_in[4];  // [128]
    const float* A_log   = (const float*)d_in[5];  // [128]
    const float* W_C     = (const float*)d_in[6];  // [1024,128]
    const float* Dvec    = (const float*)d_in[7];  // [1024]
    const float* W_out   = (const float*)d_in[8];  // [1024,1024]
    float* out = (float*)d_out;

    const int M = 4 * 2048;            // 8192 tokens
    char*  ws  = (char*)d_ws;
    size_t off = 0;
    auto alloc = [&](size_t bytes) {
        void* p = ws + off;
        off = (off + bytes + 255) & ~(size_t)255;
        return p;
    };
    unsigned short* xb   = (unsigned short*)alloc((size_t)M * 1024 * 2);
    unsigned short* Wib  = (unsigned short*)alloc((size_t)2048 * 1024 * 2);
    unsigned short* Wsd  = (unsigned short*)alloc((size_t)256 * 1024 * 2);
    unsigned short* WCb  = (unsigned short*)alloc((size_t)1024 * 128 * 2);
    unsigned short* Wob  = (unsigned short*)alloc((size_t)1024 * 1024 * 2);
    unsigned short* u    = (unsigned short*)alloc((size_t)M * 1024 * 2);
    float*          sg   = (float*)alloc((size_t)M * 1024 * 4);
    float*          bT   = (float*)alloc((size_t)M * 128 * 4);
    float*          decT = (float*)alloc((size_t)M * 128 * 4);
    unsigned short* hsb  = (unsigned short*)alloc((size_t)M * 128 * 2);
    unsigned short* z    = (unsigned short*)alloc((size_t)M * 1024 * 2);

    // ---- convert f32 operands to bf16 ----
    auto cvt = [&](const float* s, unsigned short* d, int n) {
        int blocks = (n + 255) / 256; if (blocks > 4096) blocks = 4096;
        cvt_bf16_kernel<<<blocks, 256, 0, stream>>>(s, d, n);
    };
    cvt(x,       xb,             M * 1024);
    cvt(W_in,    Wib,            2048 * 1024);
    cvt(W_state, Wsd,            128 * 1024);
    cvt(W_dt,    Wsd + 128*1024, 128 * 1024);
    cvt(W_C,     WCb,            1024 * 128);
    cvt(W_out,   Wob,            1024 * 1024);

    GemmArgs g = {};
    dim3 blk(256);

    // ---- GEMM1: ug = x @ W_in^T  (N=2048) -> u bf16, silu(gate) f32 ----
    g.A = xb; g.W = Wib; g.K = 1024; g.mode = 0;
    g.out_u = u; g.out_sg = sg;
    gemm_bf16_kernel<<<dim3(2048 / 64, M / 256), blk, 0, stream>>>(g);

    // ---- GEMM2: [b_t | dt] = u @ [W_state;W_dt]^T (N=256) -> bT, decay ----
    g = {};
    g.A = u; g.W = Wsd; g.K = 1024; g.mode = 1;
    g.out_bT = bT; g.out_dec = decT; g.b_dt = b_dt; g.A_log = A_log;
    gemm_bf16_kernel<<<dim3(256 / 64, M / 256), blk, 0, stream>>>(g);

    // ---- affine scan over time -> hs bf16 ----
    scan_kernel<<<dim3(4 * 128), blk, 0, stream>>>(decT, bT, hsb);

    // ---- GEMM3: y = hs @ W_C^T (K=128, N=1024); fuse z = y*sg + x*D ----
    g = {};
    g.A = hsb; g.W = WCb; g.K = 128; g.mode = 2;
    g.out_z = z; g.sg = sg; g.x = x; g.Dv = Dvec;
    gemm_bf16_kernel<<<dim3(1024 / 64, M / 256), blk, 0, stream>>>(g);

    // ---- GEMM4: out = z @ W_out^T (N=1024) -> f32 ----
    g = {};
    g.A = z; g.W = Wob; g.K = 1024; g.mode = 3;
    g.out_f32 = out;
    gemm_bf16_kernel<<<dim3(1024 / 64, M / 256), blk, 0, stream>>>(g);

    (void)in_sizes; (void)n_in; (void)out_size; (void)ws_size;
}
